// PriorTransformerBlock_32057635897480
// MI455X (gfx1250) — compile-verified, matrix operand order checked by reading, not run on hardware
//
#include <hip/hip_runtime.h>
#include <math.h>
#include <stdint.h>

// ---------------------------------------------------------------------------
// PriorTransformerBlock for MI455X (gfx1250, wave32, WMMA bf16).
// B=8, N=1024, D=1024, H=16, P=64, DFF=4096. ~223 GFLOP, compute bound ->
// all GEMM-like work runs on v_wmma_f32_16x16x32_bf16 (bf16 in, fp32 accum).
// LDS staging uses GLOBAL_LOAD_ASYNC_TO_LDS_B128 (ASYNCcnt) when available.
// ---------------------------------------------------------------------------

#define B_DIM   8
#define N_DIM   1024
#define D_DIM   1024
#define H_DIM   16
#define P_DIM   64
#define DFF_DIM 4096
#define SM_FACTOR 0.125f   // 1/sqrt(64)

typedef __attribute__((ext_vector_type(16))) __bf16       v16bf;
typedef __attribute__((ext_vector_type(8)))  float        v8f;
typedef __attribute__((ext_vector_type(4)))  unsigned int u32x4;
typedef int v4i __attribute__((vector_size(16)));          // <4 x i32> (builtin param type)
typedef __attribute__((address_space(1))) v4i* gas_v4i_p;  // global AS pointer
typedef __attribute__((address_space(3))) v4i* las_v4i_p;  // LDS AS pointer

union Frag { v16bf v; u32x4 q[2]; };

// ---- CDNA5 async global->LDS copy (guarded; falls back to VGPR staging) ----
#if defined(__gfx1250__) && __has_builtin(__builtin_amdgcn_global_load_async_to_lds_b128)
#define HAVE_ASYNC_LDS 1
#endif

#ifdef HAVE_ASYNC_LDS
#if __has_builtin(__builtin_amdgcn_s_wait_asynccnt)
#define WAIT_ASYNC() __builtin_amdgcn_s_wait_asynccnt(0)
#else
#define WAIT_ASYNC() asm volatile("s_wait_asynccnt 0x0" ::: "memory")
#endif
#else
#define WAIT_ASYNC() ((void)0)
#endif

// Copy 16 bytes global -> LDS. Async path writes LDS directly (no VGPR
// round-trip, tracked by ASYNCcnt); fallback is load_b128 + ds_store_b128.
// Generic->AS pointers are materialized via uintptr_t: the low 32 bits of a
// generic shared-aperture address ARE the LDS offset (ISA 10.2 aperture map).
__device__ __forceinline__ void stage_b128(const void* g, void* l) {
#ifdef HAVE_ASYNC_LDS
  __builtin_amdgcn_global_load_async_to_lds_b128(
      (gas_v4i_p)(uintptr_t)g,
      (las_v4i_p)(unsigned int)(uintptr_t)l,
      0, 0);
#else
  *(u32x4*)l = *(const u32x4*)g;
#endif
}

__device__ __forceinline__ unsigned short f2bf(float f) {
  unsigned int u = __float_as_uint(f);
  u += 0x7FFFu + ((u >> 16) & 1u);            // round-to-nearest-even
  return (unsigned short)(u >> 16);
}

// Build a 16x32 bf16 A/B fragment from an LDS row (row-major, contiguous K).
// ISA layout (CDNA5 ISA): lane<16 holds K={k0..k0+7, k0+16..k0+23},
// lane>=16 holds K={k0+8..k0+15, k0+24..k0+31}. Two ds_load_b128 per lane.
__device__ __forceinline__ v16bf load_frag(const unsigned short* rowbase, int k0, int hi) {
  Frag f;
  const unsigned short* p = rowbase + k0 + hi * 8;
  f.q[0] = *(const u32x4*)(p);
  f.q[1] = *(const u32x4*)(p + 16);
  return f.v;
}

__device__ __forceinline__ v8f wmma_bf16(v16bf a, v16bf b, v8f c) {
  // (neg_a, A, neg_b, B, c_mod, C, reuse_a, reuse_b)
  return __builtin_amdgcn_wmma_f32_16x16x32_bf16(false, a, false, b, (short)0, c, false, false);
}

__device__ __forceinline__ float gelu_f(float v) {
  const float k = 0.7978845608028654f;
  float t = tanhf(k * (v + 0.044715f * v * v * v));
  return 0.5f * v * (1.0f + t);
}

// ---------------------------------------------------------------------------
// Weight transpose + fp32->bf16 convert: W[K,N] -> Wt[N,K] (bf16).
// LDS-tiled 32x32 so both global read and write are coalesced.
// ---------------------------------------------------------------------------
__global__ __launch_bounds__(256)
void transpose_cvt_kernel(const float* __restrict__ W, unsigned short* __restrict__ Wt,
                          int K, int N) {
  __shared__ float tile[32][33];
  const int tx = threadIdx.x & 31;
  const int ty = threadIdx.x >> 5;            // 0..7
  const int nb = blockIdx.x * 32;
  const int kb = blockIdx.y * 32;
#pragma unroll
  for (int i = 0; i < 4; ++i)
    tile[ty + i * 8][tx] = W[(size_t)(kb + ty + i * 8) * N + nb + tx];
  __syncthreads();
#pragma unroll
  for (int i = 0; i < 4; ++i)
    Wt[(size_t)(nb + ty + i * 8) * K + kb + tx] = f2bf(tile[tx][ty + i * 8]);
}

// ---------------------------------------------------------------------------
// LayerNorm over D=1024, one row per 256-thread block, bf16 output.
// wave32 shfl reduce + 8-wave LDS combine.
// ---------------------------------------------------------------------------
__device__ __forceinline__ float block_reduce_sum(float v, float* tmp) {
#pragma unroll
  for (int o = 16; o > 0; o >>= 1) v += __shfl_xor(v, o, 32);
  const int wave = threadIdx.x >> 5, lane = threadIdx.x & 31;
  if (lane == 0) tmp[wave] = v;
  __syncthreads();
  float r = 0.f;
#pragma unroll
  for (int i = 0; i < 8; ++i) r += tmp[i];
  __syncthreads();
  return r;
}

__global__ __launch_bounds__(256)
void layernorm_bf16_kernel(const float* __restrict__ x, const float* __restrict__ g,
                           const float* __restrict__ bta, unsigned short* __restrict__ z) {
  __shared__ float tmp[8];
  const int row = blockIdx.x;
  const float* xr = x + (size_t)row * D_DIM;
  float v[4], s = 0.f;
#pragma unroll
  for (int i = 0; i < 4; ++i) { v[i] = xr[threadIdx.x + i * 256]; s += v[i]; }
  const float mu = block_reduce_sum(s, tmp) * (1.0f / D_DIM);
  float vs = 0.f;
#pragma unroll
  for (int i = 0; i < 4; ++i) { float d = v[i] - mu; vs += d * d; }
  const float rstd = rsqrtf(block_reduce_sum(vs, tmp) * (1.0f / D_DIM) + 1e-5f);
  unsigned short* zr = z + (size_t)row * D_DIM;
#pragma unroll
  for (int i = 0; i < 4; ++i) {
    int c = threadIdx.x + i * 256;
    zr[c] = f2bf((v[i] - mu) * rstd * g[c] + bta[c]);
  }
}

// ---------------------------------------------------------------------------
// bf16 GEMM: C = epilogue(A[M,K] @ B[K,N] + bias).  B is pre-transposed
// (Bt[N,K] bf16) so LDS staging and fragment loads are all contiguous.
// Block tile 128x128x32, 8 waves, wave tile 32x64 = 2x4 WMMA accumulators.
// Double-buffered LDS; tile k+1 streams in via async LDS loads while tile k
// runs on the WMMA pipe.
// EPI: 0 = bias -> bf16 out | 1 = bias+gelu -> bf16 out | 2 = bias+residual -> f32 out
// ---------------------------------------------------------------------------
#define GBM 128
#define GBN 128
#define GBK 32
#define GLD 40   // padded LDS row stride (ushorts), 80B = multiple of 16B

template <int EPI>
__global__ __launch_bounds__(256)
void gemm_bf16_kernel(const unsigned short* __restrict__ A,
                      const unsigned short* __restrict__ Bt,
                      const float* __restrict__ bias,
                      const float* __restrict__ residual,
                      void* __restrict__ Cout,
                      int M, int N, int K) {
  __shared__ unsigned short As[2][GBM * GLD];
  __shared__ unsigned short Bs[2][GBN * GLD];

  const int tid  = threadIdx.x;
  const int lane = tid & 31;
  const int wave = tid >> 5;
  const int wm   = wave >> 1;     // 0..3 (rows of 32)
  const int wn   = wave & 1;      // 0..1 (cols of 64)
  const int hi   = lane >> 4;
  const int l16  = lane & 15;
  const int m0   = blockIdx.y * GBM;
  const int n0   = blockIdx.x * GBN;

  v8f acc[2][4] = {};

  auto stage = [&](int buf, int k0) {
#pragma unroll
    for (int s = 0; s < 2; ++s) {
      int c   = tid + s * 256;              // 512 chunks of 8 ushorts
      int row = c >> 2;
      int kc  = (c & 3) << 3;
      stage_b128(&A [(size_t)(m0 + row) * K + k0 + kc], &As[buf][row * GLD + kc]);
      stage_b128(&Bt[(size_t)(n0 + row) * K + k0 + kc], &Bs[buf][row * GLD + kc]);
    }
  };

  stage(0, 0);
  WAIT_ASYNC();
  __syncthreads();

  const int nk = K / GBK;
  for (int kt = 0; kt < nk; ++kt) {
    const int buf = kt & 1;
    if (kt + 1 < nk) stage(buf ^ 1, (kt + 1) * GBK);
    if (kt + 2 < nk) {  // streaming prefetch -> global_prefetch_b8
      __builtin_prefetch(&A [(size_t)(m0 + (tid >> 1)) * K + (kt + 2) * GBK], 0, 0);
      __builtin_prefetch(&Bt[(size_t)(n0 + (tid >> 1)) * K + (kt + 2) * GBK], 0, 0);
    }

    v16bf af[2], bfv[4];
#pragma unroll
    for (int mi = 0; mi < 2; ++mi)
      af[mi] = load_frag(&As[buf][(wm * 32 + mi * 16 + l16) * GLD], 0, hi);
#pragma unroll
    for (int ni = 0; ni < 4; ++ni)
      bfv[ni] = load_frag(&Bs[buf][(wn * 64 + ni * 16 + l16) * GLD], 0, hi);
#pragma unroll
    for (int mi = 0; mi < 2; ++mi)
#pragma unroll
      for (int ni = 0; ni < 4; ++ni)
        acc[mi][ni] = wmma_bf16(af[mi], bfv[ni], acc[mi][ni]);
    WAIT_ASYNC();
    __syncthreads();
  }

  // Epilogue. C layout: VGPR r -> M = r + 8*hi, N = l16.
#pragma unroll
  for (int mi = 0; mi < 2; ++mi)
#pragma unroll
    for (int ni = 0; ni < 4; ++ni) {
      const int col = n0 + wn * 64 + ni * 16 + l16;
      const float bv = bias[col];
#pragma unroll
      for (int r = 0; r < 8; ++r) {
        const int row = m0 + wm * 32 + mi * 16 + r + hi * 8;
        float v = acc[mi][ni][r] + bv;
        const size_t idx = (size_t)row * N + col;
        if (EPI == 1) v = gelu_f(v);
        if (EPI == 2) ((float*)Cout)[idx] = residual[idx] + v;
        else          ((unsigned short*)Cout)[idx] = f2bf(v);
      }
    }
}

// ---------------------------------------------------------------------------
// Flash attention: one block per (b,h, 128-query tile). 256 threads, 8 waves.
// S = Q.K^T (WMMA) -> LDS fp32; fused +alpha*prior, *1/sqrt(P); online
// softmax with running row max/sum; O += P(bf16).V (WMMA); epilogue fuses
// residual x. Dynamic LDS ~158KB (CDNA5 WGP has 320KB).
// ---------------------------------------------------------------------------
#define AQ_LD 72    // Q/K LDS row stride (ushorts); 144B multiple of 16
#define AV_LD 136   // V^T LDS row stride
#define AP_LD 136   // P LDS row stride
#define AS_LD 132   // S fp32 LDS row stride; 528B multiple of 16

__global__ __launch_bounds__(256)
void attn_kernel(const unsigned short* __restrict__ qkv,   // [B*N, 3D] bf16
                 const float* __restrict__ x,              // [B,N,D]
                 const float* __restrict__ prior,          // [B,N,N]
                 const float* __restrict__ alpha_p,
                 float* __restrict__ s1)                   // [B,N,D] = x + attn out
{
  extern __shared__ char smem_raw[];
  unsigned short* Qs  = (unsigned short*)smem_raw;     // 128*72
  unsigned short* Ks  = Qs  + 128 * AQ_LD;             // 128*72
  unsigned short* Vts = Ks  + 128 * AQ_LD;             //  64*136 (V transposed)
  unsigned short* Ps  = Vts + P_DIM * AV_LD;           // 128*136
  float* Ss       = (float*)(Ps + 128 * AP_LD);        // 128*132 fp32
  float* rowm     = Ss + 128 * AS_LD;
  float* rowl     = rowm + 128;
  float* rowscale = rowl + 128;

  const int tid  = threadIdx.x;
  const int lane = tid & 31;
  const int wave = tid >> 5;
  const int wm   = wave >> 1;   // 0..3
  const int wn   = wave & 1;    // 0..1
  const int hi   = lane >> 4;
  const int l16  = lane & 15;

  const int i0 = blockIdx.x * 128;
  const int bh = blockIdx.y;
  const int b  = bh >> 4;
  const int h  = bh & 15;
  const int qkv_ld = 3 * D_DIM;
  const float alpha = alpha_p[0];

  // Stage Q tile [128 x 64] bf16 (stays resident for all j tiles) — async.
  const size_t qbase = ((size_t)b * N_DIM + i0) * qkv_ld + h * P_DIM;
#pragma unroll
  for (int s = 0; s < 4; ++s) {
    int c = tid + s * 256; int row = c >> 3; int cc = (c & 7) << 3;
    stage_b128(&qkv[qbase + (size_t)row * qkv_ld + cc], &Qs[row * AQ_LD + cc]);
  }
  if (tid < 128) { rowm[tid] = -1e30f; rowl[tid] = 0.f; }

  v8f o[2][2] = {};   // O accumulators: wave tile 32 rows x 32 head-dims

  for (int jt = 0; jt < N_DIM / 128; ++jt) {
    const int j0 = jt * 128;
    // Stage K tile [128 x 64] — async.
    const size_t kbase = ((size_t)b * N_DIM + j0) * qkv_ld + D_DIM + h * P_DIM;
#pragma unroll
    for (int s = 0; s < 4; ++s) {
      int c = tid + s * 256; int row = c >> 3; int cc = (c & 7) << 3;
      stage_b128(&qkv[kbase + (size_t)row * qkv_ld + cc], &Ks[row * AQ_LD + cc]);
    }
    // Stage V tile transposed: Vts[p][j] (needs element shuffle -> VGPR path).
    const size_t vbase = ((size_t)b * N_DIM + j0) * qkv_ld + 2 * D_DIM + h * P_DIM;
#pragma unroll
    for (int s = 0; s < 4; ++s) {
      int c = tid + s * 256; int row = c >> 3; int cc = (c & 7) << 3;
      u32x4 d = *(const u32x4*)&qkv[vbase + (size_t)row * qkv_ld + cc];
      const unsigned short* e = (const unsigned short*)&d;
#pragma unroll
      for (int t = 0; t < 8; ++t) Vts[(cc + t) * AV_LD + row] = e[t];
    }
    WAIT_ASYNC();
    __syncthreads();

    // S = Q.K^T : wave tile 32x64, K-dim = 64 (2 WMMA steps).
    v8f sacc[2][4] = {};
#pragma unroll
    for (int k0 = 0; k0 < P_DIM; k0 += 32) {
      v16bf af[2], bfv[4];
#pragma unroll
      for (int mi = 0; mi < 2; ++mi)
        af[mi] = load_frag(&Qs[(wm * 32 + mi * 16 + l16) * AQ_LD], k0, hi);
#pragma unroll
      for (int ni = 0; ni < 4; ++ni)
        bfv[ni] = load_frag(&Ks[(wn * 64 + ni * 16 + l16) * AQ_LD], k0, hi);
#pragma unroll
      for (int mi = 0; mi < 2; ++mi)
#pragma unroll
        for (int ni = 0; ni < 4; ++ni)
          sacc[mi][ni] = wmma_bf16(af[mi], bfv[ni], sacc[mi][ni]);
    }
#pragma unroll
    for (int mi = 0; mi < 2; ++mi)
#pragma unroll
      for (int ni = 0; ni < 4; ++ni) {
        const int col = wn * 64 + ni * 16 + l16;
#pragma unroll
        for (int r = 0; r < 8; ++r)
          Ss[(wm * 32 + mi * 16 + r + hi * 8) * AS_LD + col] = sacc[mi][ni][r];
      }
    __syncthreads();

    // Online softmax: two threads per row (64 cols each), fused prior add.
    {
      const int r = tid >> 1, half = tid & 1, c0 = half * 64;
      const float* pr = &prior[((size_t)b * N_DIM + i0 + r) * N_DIM + j0 + c0];
      float* srow = &Ss[r * AS_LD + c0];
      float pmax = -1e30f;
#pragma unroll 8
      for (int c = 0; c < 64; ++c) {
        float v = (srow[c] + alpha * pr[c]) * SM_FACTOR;
        srow[c] = v;
        pmax = fmaxf(pmax, v);
      }
      pmax = fmaxf(pmax, __shfl_xor(pmax, 1, 32));
      const float mold = rowm[r];
      const float mnew = fmaxf(mold, pmax);
      const float scl  = __expf(mold - mnew);
      unsigned short* prow = &Ps[r * AP_LD + c0];
      float psum = 0.f;
#pragma unroll 8
      for (int c = 0; c < 64; ++c) {
        float p = __expf(srow[c] - mnew);
        prow[c] = f2bf(p);
        psum += p;
      }
      psum += __shfl_xor(psum, 1, 32);
      if (half == 0) {
        rowm[r] = mnew;
        rowl[r] = rowl[r] * scl + psum;
        rowscale[r] = scl;
      }
    }
    __syncthreads();

    // Rescale running O, then O += P.V (K-dim = 128, 4 WMMA steps).
#pragma unroll
    for (int mi = 0; mi < 2; ++mi)
#pragma unroll
      for (int ni = 0; ni < 2; ++ni)
#pragma unroll
        for (int r = 0; r < 8; ++r)
          o[mi][ni][r] *= rowscale[wm * 32 + mi * 16 + r + hi * 8];
#pragma unroll
    for (int k0 = 0; k0 < 128; k0 += 32) {
      v16bf af[2], bfv[2];
#pragma unroll
      for (int mi = 0; mi < 2; ++mi)
        af[mi] = load_frag(&Ps[(wm * 32 + mi * 16 + l16) * AP_LD], k0, hi);
#pragma unroll
      for (int ni = 0; ni < 2; ++ni)
        bfv[ni] = load_frag(&Vts[(wn * 32 + ni * 16 + l16) * AV_LD], k0, hi);
#pragma unroll
      for (int mi = 0; mi < 2; ++mi)
#pragma unroll
        for (int ni = 0; ni < 2; ++ni)
          o[mi][ni] = wmma_bf16(af[mi], bfv[ni], o[mi][ni]);
    }
    __syncthreads();
  }

  // Epilogue: s1 = x + O / l
#pragma unroll
  for (int mi = 0; mi < 2; ++mi)
#pragma unroll
    for (int ni = 0; ni < 2; ++ni) {
      const int colp = wn * 32 + ni * 16 + l16;
#pragma unroll
      for (int r = 0; r < 8; ++r) {
        const int row = wm * 32 + mi * 16 + r + hi * 8;
        const float inv = 1.0f / rowl[row];
        const size_t gi = ((size_t)b * N_DIM + i0 + row) * D_DIM + h * P_DIM + colp;
        s1[gi] = x[gi] + o[mi][ni][r] * inv;
      }
    }
}

// ---------------------------------------------------------------------------
// Host orchestration
// ---------------------------------------------------------------------------
extern "C" void kernel_launch(void* const* d_in, const int* in_sizes, int n_in,
                              void* d_out, int out_size, void* d_ws, size_t ws_size,
                              hipStream_t stream) {
  const float* x     = (const float*)d_in[0];
  const float* prior = (const float*)d_in[1];
  const float* Wqkv  = (const float*)d_in[2];
  const float* bqkv  = (const float*)d_in[3];
  const float* alpha = (const float*)d_in[4];
  const float* ln1g  = (const float*)d_in[5];
  const float* ln1b  = (const float*)d_in[6];
  const float* ln2g  = (const float*)d_in[7];
  const float* ln2b  = (const float*)d_in[8];
  const float* w1    = (const float*)d_in[9];
  const float* b1    = (const float*)d_in[10];
  const float* w2    = (const float*)d_in[11];
  const float* b2    = (const float*)d_in[12];
  float* out = (float*)d_out;

  const size_t R = (size_t)B_DIM * N_DIM;   // 8192 token rows
  size_t off = 0;
  auto carve = [&](size_t bytes) {
    void* p = (char*)d_ws + off;
    off += (bytes + 255) & ~(size_t)255;
    return p;
  };
  unsigned short* z1    = (unsigned short*)carve(R * D_DIM * 2);
  unsigned short* z2    = (unsigned short*)carve(R * D_DIM * 2);
  unsigned short* qkvb  = (unsigned short*)carve(R * 3 * D_DIM * 2);
  unsigned short* hmid  = (unsigned short*)carve(R * DFF_DIM * 2);
  unsigned short* WqkvT = (unsigned short*)carve((size_t)3 * D_DIM * D_DIM * 2);
  unsigned short* w1T   = (unsigned short*)carve((size_t)DFF_DIM * D_DIM * 2);
  unsigned short* w2T   = (unsigned short*)carve((size_t)D_DIM * DFF_DIM * 2);
  float*          s1w   = (float*)carve(R * D_DIM * 4);

  const dim3 tb(256);

  // Weights -> bf16 transposed (Bt layout for the GEMMs)
  transpose_cvt_kernel<<<dim3(3 * D_DIM / 32, D_DIM / 32),  tb, 0, stream>>>(Wqkv, WqkvT, D_DIM,   3 * D_DIM);
  transpose_cvt_kernel<<<dim3(DFF_DIM / 32,  D_DIM / 32),   tb, 0, stream>>>(w1,   w1T,   D_DIM,   DFF_DIM);
  transpose_cvt_kernel<<<dim3(D_DIM / 32,    DFF_DIM / 32), tb, 0, stream>>>(w2,   w2T,   DFF_DIM, D_DIM);

  // z1 = LN1(x) (bf16)
  layernorm_bf16_kernel<<<(int)R, tb, 0, stream>>>(x, ln1g, ln1b, z1);

  // qkv = z1 @ Wqkv + bqkv  (bf16 out)
  gemm_bf16_kernel<0><<<dim3(3 * D_DIM / GBN, R / GBM), tb, 0, stream>>>(
      z1, WqkvT, bqkv, nullptr, qkvb, (int)R, 3 * D_DIM, D_DIM);

  // s1 = x + softmax((q k^T + a*prior)/sqrt(P)) v   (flash, fused residual)
  const size_t attn_smem =
      (size_t)(128 * AQ_LD + 128 * AQ_LD + P_DIM * AV_LD + 128 * AP_LD) * 2 +
      (size_t)(128 * AS_LD + 3 * 128) * 4;   // ~158 KB < 320 KB WGP LDS
  attn_kernel<<<dim3(N_DIM / 128, B_DIM * H_DIM), tb, attn_smem, stream>>>(
      qkvb, x, prior, alpha, s1w);

  // z2 = LN2(s1) (bf16)
  layernorm_bf16_kernel<<<(int)R, tb, 0, stream>>>(s1w, ln2g, ln2b, z2);

  // hmid = gelu(z2 @ w1 + b1) (bf16)
  gemm_bf16_kernel<1><<<dim3(DFF_DIM / GBN, R / GBM), tb, 0, stream>>>(
      z2, w1T, b1, nullptr, hmid, (int)R, DFF_DIM, D_DIM);

  // out = s1 + hmid @ w2 + b2 (fp32)
  gemm_bf16_kernel<2><<<dim3(D_DIM / GBN, R / GBM), tb, 0, stream>>>(
      hmid, w2T, b2, s1w, out, (int)R, D_DIM, DFF_DIM);
}
